// VoxAttentionL2_58016418235044
// MI455X (gfx1250) — matrix operand order repaired and checked by reading, not compiled, not run on hardware
//
#include <hip/hip_runtime.h>
#include <hip/hip_bf16.h>

// ---------------- constants ----------------
#define NP   4096      // patches
#define ED   64        // embed dim E
#define NH   8         // heads
#define HD   512       // H*E
#define EPS  1e-5f
#define SCALE 0.125f   // 1/sqrt(64)
#define VSTR 40        // padded stride (halves) of transposed V tile: 80B rows, 16B aligned

#define USE_ASYNC 1    // gfx1250 GLOBAL_LOAD_ASYNC_TO_LDS path for the K tile

typedef __attribute__((ext_vector_type(16))) _Float16 v16h;
typedef __attribute__((ext_vector_type(8)))  _Float16 v8h;
typedef __attribute__((ext_vector_type(8)))  float    v8f;

union AFrag { v16h v; v8h q[2]; _Float16 h[16]; };

static __device__ __forceinline__ v8f wmma_f16(v16h a, v16h b, v8f c) {
    return __builtin_amdgcn_wmma_f32_16x16x32_f16(
        false, a, false, b, (short)0, c, false, false);
}

// ---------------- kernel 1: patch embedding + LayerNorm + pos ----------------
__global__ void __launch_bounds__(64)
k_embed(const float* __restrict__ voxel, const float* __restrict__ conv_w,
        const float* __restrict__ conv_b, const float* __restrict__ ln_g,
        const float* __restrict__ ln_b, const float* __restrict__ pos,
        float* __restrict__ emb) {
    __shared__ float patch[64];
    __shared__ float erow[64];
    const int n = blockIdx.x;
    const int t = threadIdx.x;
    const int pz = n >> 8, py = (n >> 4) & 15, px = n & 15;
    {
        const int dz = t >> 4, dy = (t >> 2) & 3, dx = t & 3;
        patch[t] = voxel[((pz * 4 + dz) * 64 + (py * 4 + dy)) * 64 + (px * 4 + dx)];
    }
    __syncthreads();
    float acc = conv_b[t];
    #pragma unroll 8
    for (int i = 0; i < 64; ++i) acc += patch[i] * conv_w[t * 64 + i];
    erow[t] = acc;
    __syncthreads();
    float mu = 0.f;
    #pragma unroll 8
    for (int i = 0; i < 64; ++i) mu += erow[i];
    mu *= (1.f / 64.f);
    float var = 0.f;
    #pragma unroll 8
    for (int i = 0; i < 64; ++i) { float d = erow[i] - mu; var += d * d; }
    var *= (1.f / 64.f);
    float x = (acc - mu) * rsqrtf(var + EPS) * ln_g[t] + ln_b[t] + pos[n * 64 + t];
    emb[n * 64 + t] = x;
}

// ---------------- kernel 2: QK / V projections + per-head |q|^2 ----------------
__global__ void __launch_bounds__(128)
k_proj(const float* __restrict__ emb, const float* __restrict__ wqk,
       const float* __restrict__ bqk, const float* __restrict__ wv,
       const float* __restrict__ bv, _Float16* __restrict__ qkh,
       _Float16* __restrict__ vh, float* __restrict__ q2) {
    __shared__ float row[64];
    __shared__ float qkr[512];
    const int n = blockIdx.x, t = threadIdx.x;
    if (t < 64) row[t] = emb[n * 64 + t];
    __syncthreads();
    for (int c = t; c < 512; c += 128) {
        float a = bqk[c];
        float av = bv[c];
        #pragma unroll 8
        for (int e = 0; e < 64; ++e) {
            float r = row[e];
            a  += r * wqk[e * 512 + c];
            av += r * wv [e * 512 + c];
        }
        const int h = c >> 6, eo = c & 63;
        _Float16 a16 = (_Float16)a;
        qkh[(h * NP + n) * 64 + eo] = a16;
        vh [(h * NP + n) * 64 + eo] = (_Float16)av;
        qkr[c] = (float)a16;
    }
    __syncthreads();
    if (t < NH) {
        float s = 0.f;
        #pragma unroll 8
        for (int e = 0; e < 64; ++e) { float q = qkr[t * 64 + e]; s += q * q; }
        q2[t * NP + n] = s;
    }
}

// ---------------- kernel 3: L2-distance flash attention (WMMA) ----------------
// grid = (64 row-blocks, 8 heads), 128 threads = 4 waves; wave w owns rows rb*64+w*16..+15.
// K tile: async global->LDS copy (row-major). V tile: transposed into LDS so B-fragment
// reads are contiguous ds_load_b128. Scores bounded in (0, alpha/8] -> single-pass softmax.
__global__ void __launch_bounds__(128)
k_attn(const _Float16* __restrict__ qkh, const _Float16* __restrict__ vhm,
       const float* __restrict__ q2, const float* __restrict__ alpha_p,
       float* __restrict__ attn) {
    __shared__ _Float16 kT[32 * 64];         // K tile row-major [32 k][64 e]
    __shared__ _Float16 vTt[64 * VSTR];      // V tile transposed [64 e][32 k], padded
    __shared__ _Float16 pT[4][16 * 32];      // per-wave P tile [16 rows][32 cols]

    const int h    = blockIdx.y;
    const int rb   = blockIdx.x;
    const int t    = threadIdx.x;
    const int wave = t >> 5;
    const int lane = t & 31;
    const int lo16 = lane & 15;
    const int hi   = lane >> 4;

    const _Float16* qh  = qkh + (size_t)h * NP * 64;
    const _Float16* vh  = vhm + (size_t)h * NP * 64;
    const float*    q2h = q2  + (size_t)h * NP;
    const int row0 = rb * 64 + wave * 16;
    const float aS = alpha_p[0] * SCALE;

#if USE_ASYNC
    const unsigned kT_lds = (unsigned)(uintptr_t)(void*)&kT[0];
#endif

    // ---- preload Q A-fragments (2 K-chunks of 32 dims) ----
    AFrag qa[2];
    {
        const int r = row0 + lo16;
        const int kb = hi * 16;
        #pragma unroll
        for (int ch = 0; ch < 2; ++ch) {
            qa[ch].q[0] = *(const v8h*)(qh + (size_t)r * 64 + ch * 32 + kb);
            qa[ch].q[1] = *(const v8h*)(qh + (size_t)r * 64 + ch * 32 + kb + 8);
        }
    }
    float q2row[8];
    #pragma unroll
    for (int r = 0; r < 8; ++r) q2row[r] = q2h[row0 + r + 8 * hi];

    v8f o[4] = {};
    float rsum[8];
    #pragma unroll
    for (int r = 0; r < 8; ++r) rsum[r] = 0.f;

    _Float16* pw = &pT[wave][0];

    for (int ct = 0; ct < NP; ct += 32) {
        __syncthreads();   // previous iteration done reading tiles

        // ---- K tile: 4096 B, 128 threads x 32 B (2x B128 per thread) ----
        const _Float16* kg = qh + (size_t)ct * 64 + t * 8;
#if USE_ASYNC
        {
            unsigned d0 = kT_lds + (unsigned)t * 16u;
            asm volatile("global_load_async_to_lds_b128 %0, %1, off"
                         :: "v"(d0), "v"(kg) : "memory");
            asm volatile("global_load_async_to_lds_b128 %0, %1, off"
                         :: "v"(d0 + 2048u), "v"(kg + 1024) : "memory");
        }
#else
        *(uint4*)&kT[t * 8]        = *(const uint4*)(kg);
        *(uint4*)&kT[t * 8 + 1024] = *(const uint4*)(kg + 1024);
#endif

        // ---- V tile: read 32 B/thread from global, scatter transposed into LDS ----
        {
            const _Float16* vg = vh + (size_t)ct * 64 + t * 16;
            AFrag tmp;
            tmp.q[0] = *(const v8h*)(vg);
            tmp.q[1] = *(const v8h*)(vg + 8);
            const int k  = (t * 16) >> 6;     // tile row (constant over the 16 elems)
            const int e0 = (t * 16) & 63;
            #pragma unroll
            for (int i = 0; i < 16; ++i)
                vTt[(e0 + i) * VSTR + k] = tmp.h[i];
        }
#if USE_ASYNC
        asm volatile("s_wait_asynccnt 0x0" ::: "memory");
#endif
        __syncthreads();

        // ---- S = Q K^T for cols [ct, ct+32): two 16x16 accs, 2 WMMAs each ----
        v8f s0 = {}, s1 = {};
        #pragma unroll
        for (int ch = 0; ch < 2; ++ch) {
            v16h b0 = *(const v16h*)&kT[lo16 * 64        + ch * 32 + hi * 16];
            v16h b1 = *(const v16h*)&kT[(16 + lo16) * 64 + ch * 32 + hi * 16];
            s0 = wmma_f16(qa[ch].v, b0, s0);
            s1 = wmma_f16(qa[ch].v, b1, s1);
        }

        // ---- probabilities: p = exp(alpha*SCALE*exp(-dist)) ----
        const float q2c0 = q2h[ct + lo16];
        const float q2c1 = q2h[ct + 16 + lo16];
        #pragma unroll
        for (int r = 0; r < 8; ++r) {
            float d0 = q2row[r] + q2c0 - 2.f * s0[r];
            float d1 = q2row[r] + q2c1 - 2.f * s1[r];
            float p0 = __expf(aS * __expf(-d0));
            float p1 = __expf(aS * __expf(-d1));
            rsum[r] += p0 + p1;
            const int lr = r + 8 * hi;
            pw[lr * 32 + lo16]      = (_Float16)p0;
            pw[lr * 32 + 16 + lo16] = (_Float16)p1;
        }
        // intra-wave LDS ops are in-order: reads below see this wave's writes above.

        // ---- O += P (16x32) x V (32x64): 4 WMMAs, all B-frags contiguous ----
        AFrag pa;
        pa.q[0] = *(const v8h*)&pw[lo16 * 32 + hi * 16];
        pa.q[1] = *(const v8h*)&pw[lo16 * 32 + hi * 16 + 8];
        #pragma unroll
        for (int j = 0; j < 4; ++j) {
            v16h vb = *(const v16h*)&vTt[(j * 16 + lo16) * VSTR + hi * 16];
            o[j] = wmma_f16(pa.v, vb, o[j]);
        }
    }

    // ---- reduce rowsums across the 16 lanes sharing each row ----
    #pragma unroll
    for (int r = 0; r < 8; ++r)
        #pragma unroll
        for (int m = 1; m < 16; m <<= 1)
            rsum[r] += __shfl_xor(rsum[r], m, 32);

    // ---- normalize + store attn[n][h*64+e] ----
    #pragma unroll
    for (int r = 0; r < 8; ++r) {
        const float inv = 1.f / rsum[r];
        const int n = row0 + r + 8 * hi;
        #pragma unroll
        for (int j = 0; j < 4; ++j)
            attn[(size_t)n * HD + h * 64 + j * 16 + lo16] = o[j][r] * inv;
    }
}

// ---------------- kernel 4: output projection [N,512] x [512,64] + bias ----------------
__global__ void __launch_bounds__(64)
k_out(const float* __restrict__ attn, const float* __restrict__ wout,
      const float* __restrict__ bout, float* __restrict__ out) {
    __shared__ float row[512];
    const int n = blockIdx.x, t = threadIdx.x;
    for (int i = t; i < 512; i += 64) row[i] = attn[(size_t)n * HD + i];
    __syncthreads();
    float a = bout[t];
    #pragma unroll 8
    for (int k = 0; k < 512; ++k) a += row[k] * wout[k * 64 + t];
    out[n * 64 + t] = a;
}

// ---------------- host launcher ----------------
extern "C" void kernel_launch(void* const* d_in, const int* in_sizes, int n_in,
                              void* d_out, int out_size, void* d_ws, size_t ws_size,
                              hipStream_t stream) {
    const float* voxel  = (const float*)d_in[0];
    const float* conv_w = (const float*)d_in[1];
    const float* conv_b = (const float*)d_in[2];
    const float* ln_g   = (const float*)d_in[3];
    const float* ln_b   = (const float*)d_in[4];
    const float* pos    = (const float*)d_in[5];
    const float* wqk    = (const float*)d_in[6];
    const float* bqk    = (const float*)d_in[7];
    const float* wv     = (const float*)d_in[8];
    const float* bv     = (const float*)d_in[9];
    const float* alpha  = (const float*)d_in[10];
    const float* wout   = (const float*)d_in[11];
    const float* bout   = (const float*)d_in[12];
    float* out = (float*)d_out;

    char* ws = (char*)d_ws;
    float*     emb  = (float*)ws;    ws += (size_t)NP * ED * sizeof(float);
    _Float16*  qkh  = (_Float16*)ws; ws += (size_t)NH * NP * ED * sizeof(_Float16);
    _Float16*  vh   = (_Float16*)ws; ws += (size_t)NH * NP * ED * sizeof(_Float16);
    float*     q2   = (float*)ws;    ws += (size_t)NH * NP * sizeof(float);
    float*     attn = (float*)ws;

    k_embed<<<NP, 64, 0, stream>>>(voxel, conv_w, conv_b, ln_g, ln_b, pos, emb);
    k_proj <<<NP, 128, 0, stream>>>(emb, wqk, bqk, wv, bv, qkh, vh, q2);
    k_attn <<<dim3(NP / 64, NH), 128, 0, stream>>>(qkh, vh, q2, alpha, attn);
    k_out  <<<NP, 64, 0, stream>>>(attn, wout, bout, out);
}
